// AgentRoadGraphAttention_31774168056199
// MI455X (gfx1250) — matrix operand order repaired and checked by reading, not compiled, not run on hardware
//
#include <hip/hip_runtime.h>

// ---------------------------------------------------------------------------
// Problem constants (from reference)
// ---------------------------------------------------------------------------
#define BB   2
#define AA   64
#define TT   80
#define FF   256      // feature dim = D_K
#define RG   1024
#define NH   4
#define DH   64
#define MAT  (AA * TT)        // 5120 query rows per batch
#define MALL (BB * MAT)       // 10240 query rows total
#define MKV  (BB * RG)        // 2048 kv rows total

#define DEVFN __device__ __forceinline__

typedef __attribute__((ext_vector_type(8)))  float  v8f;
typedef __attribute__((ext_vector_type(16))) __bf16 v16bf;

union FragCast { int4 q[2]; v16bf f; };

DEVFN unsigned short f2bf(float x) {
  unsigned int u = __float_as_uint(x);
  u += 0x7FFFu + ((u >> 16) & 1u);          // round-to-nearest-even
  return (unsigned short)(u >> 16);
}

// A-fragment (16x32 bf16, M x K). base points at element (row0, k0), ldk in elems.
// Lanes 0-15: M=lane, K 0..15; lanes 16-31: M=lane-16, K 16..31.
DEVFN v16bf load_a_frag(const unsigned short* base, int ldk, int lane) {
  const int half = lane >> 4;
  const unsigned short* p = base + (size_t)(lane & 15) * ldk + half * 16;
  FragCast u;
  u.q[0] = *(const int4*)(p);
  u.q[1] = *(const int4*)(p + 8);
  return u.f;
}

// B-fragment (32x16 bf16, K x N) where B[k][n] = base[n*ldk + k] (k-contiguous).
// Lanes 0-15: N=lane, K 0..15; lanes 16-31: N=lane-16, K 16..31. Dword j = K pair (2j,2j+1).
DEVFN v16bf load_b_frag(const unsigned short* base, int ldk, int lane) {
  const unsigned short* p = base + (size_t)(lane & 15) * ldk + (lane >> 4) * 16;
  FragCast u;
  u.q[0] = *(const int4*)(p);
  u.q[1] = *(const int4*)(p + 8);
  return u.f;
}

DEVFN v8f wmma_bf16(v16bf a, v16bf b, v8f c) {
  return __builtin_amdgcn_wmma_f32_16x16x32_bf16(false, a, false, b, (short)0, c,
                                                 false, false);
}

// ---------------------------------------------------------------------------
// fp32 -> bf16 conversion (weights, roadgraph)
// ---------------------------------------------------------------------------
__global__ void cvt_bf16_kernel(const float* __restrict__ src,
                                unsigned short* __restrict__ dst, int n) {
  for (int i = blockIdx.x * blockDim.x + threadIdx.x; i < n;
       i += gridDim.x * blockDim.x)
    dst[i] = f2bf(src[i]);
}

// per-batch any(mask)
__global__ void mask_any_kernel(const int* __restrict__ mask,
                                int* __restrict__ flags) {
  const int b = blockIdx.x;
  int v = 0;
  for (int i = threadIdx.x; i < RG; i += blockDim.x) v |= mask[b * RG + i];
#pragma unroll
  for (int st = 16; st >= 1; st >>= 1) v |= __shfl_xor(v, st, 32);
  __shared__ int sm[8];
  if ((threadIdx.x & 31) == 0) sm[threadIdx.x >> 5] = v;
  __syncthreads();
  if (threadIdx.x == 0) {
    int r = 0;
#pragma unroll
    for (int i = 0; i < 8; i++) r |= sm[i];
    flags[b] = (r != 0) ? 1 : 0;
  }
}

// ---------------------------------------------------------------------------
// LayerNorm: one 256-thread block per row; writes fp32 xn (residual base,
// into d_out) and bf16 xn (GEMM operand, into ws)
// ---------------------------------------------------------------------------
__global__ void ln_kernel(const float* __restrict__ x,
                          const float* __restrict__ gamma,
                          const float* __restrict__ beta,
                          float* __restrict__ xn,
                          unsigned short* __restrict__ xnbf) {
  const int row = blockIdx.x;
  const int tid = threadIdx.x;
  const float v = x[(size_t)row * FF + tid];
  float s = v, s2 = v * v;
#pragma unroll
  for (int st = 16; st >= 1; st >>= 1) {
    s  += __shfl_xor(s, st, 32);
    s2 += __shfl_xor(s2, st, 32);
  }
  __shared__ float rs[8], rs2[8];
  if ((tid & 31) == 0) { rs[tid >> 5] = s; rs2[tid >> 5] = s2; }
  __syncthreads();
  float ts = 0.f, ts2 = 0.f;
#pragma unroll
  for (int i = 0; i < 8; i++) { ts += rs[i]; ts2 += rs2[i]; }
  const float mu  = ts * (1.f / FF);
  const float var = ts2 * (1.f / FF) - mu * mu;
  const float inv = rsqrtf(var + 1e-5f);
  const float y = (v - mu) * inv * gamma[tid] + beta[tid];
  xn[(size_t)row * FF + tid]   = y;
  xnbf[(size_t)row * FF + tid] = f2bf(y);
}

// ---------------------------------------------------------------------------
// Projection GEMM: out[m,n] = sum_k A[m,k] * W[n,k] + bias[n]
// Block = 4 waves, each wave a 16x64 tile (4 WMMA accumulators).
// vmode 0: out[((b*NH+h)*Mb + r)*DH + d]   (Q, K: row-major per head)
// vmode 1: out[((b*NH+h)*DH + d)*Mb + r]   (V: transposed per head)
// ---------------------------------------------------------------------------
__global__ void proj_gemm_kernel(const unsigned short* __restrict__ Abf,
                                 const unsigned short* __restrict__ Wbf,
                                 const float* __restrict__ bias,
                                 unsigned short* __restrict__ outbf,
                                 int Mb, int vmode) {
  const int lane = threadIdx.x & 31;
  const int wv   = threadIdx.x >> 5;
  const int m0   = blockIdx.y * 64 + wv * 16;
  const int n0   = blockIdx.x * 64;
  v8f zero = {};
  v8f acc[4];
#pragma unroll
  for (int t = 0; t < 4; t++) acc[t] = zero;

  for (int k0 = 0; k0 < FF; k0 += 32) {
    v16bf a = load_a_frag(Abf + (size_t)m0 * FF + k0, FF, lane);
#pragma unroll
    for (int t = 0; t < 4; t++) {
      v16bf bfrag = load_b_frag(Wbf + (size_t)(n0 + t * 16) * FF + k0, FF, lane);
      acc[t] = wmma_bf16(a, bfrag, acc[t]);
    }
  }

  const int nl = lane & 15;
  const int mh = (lane >> 4) * 8;
#pragma unroll
  for (int t = 0; t < 4; t++) {
    const int n = n0 + t * 16 + nl;
    const float bn = bias[n];
    const int h = n >> 6, d = n & 63;
#pragma unroll
    for (int v = 0; v < 8; v++) {
      const int m = m0 + mh + v;
      const int b = m / Mb;
      const int r = m - b * Mb;
      const float val = acc[t][v] + bn;
      size_t idx = vmode
          ? ((size_t)((b * NH + h) * DH + d) * (size_t)Mb + r)
          : ((size_t)((b * NH + h) * Mb + r) * (size_t)DH + d);
      outbf[idx] = f2bf(val);
    }
  }
}

// ---------------------------------------------------------------------------
// Flash attention: per wave one 16-query tile, streaming RG in chunks of 32.
// Q: (b,h, MAT, DH) bf16 row-major; K: (b,h, RG, DH) bf16 row-major;
// Vt: (b,h, DH, RG) bf16 (transposed). Out: (b, MAT, NH*DH) bf16.
// ---------------------------------------------------------------------------
__global__ void attn_kernel(const unsigned short* __restrict__ Qw,
                            const unsigned short* __restrict__ Kw,
                            const unsigned short* __restrict__ Vt,
                            const int* __restrict__ mask,
                            const int* __restrict__ flags,
                            unsigned short* __restrict__ attnout) {
  const int lane = threadIdx.x & 31;
  const int wv   = threadIdx.x >> 5;
  const int b    = blockIdx.z;
  const int h    = blockIdx.y;
  const int bh   = b * NH + h;
  const int m0   = blockIdx.x * 64 + wv * 16;   // query row within batch

  const unsigned short* Qb = Qw + ((size_t)bh * MAT + m0) * DH;
  const unsigned short* Kb = Kw + (size_t)bh * RG * DH;
  const unsigned short* Vb = Vt + (size_t)bh * DH * RG;
  const int* mb = mask + (size_t)b * RG;

  __shared__ __align__(16) unsigned short sP[4][16 * 32];
  unsigned short* myP = sP[wv];

  // Q fragments: reused across the whole RG loop.
  v16bf qf0 = load_a_frag(Qb, DH, lane);        // d 0..31
  v16bf qf1 = load_a_frag(Qb + 32, DH, lane);   // d 32..63

  v8f zero = {};
  v8f o0 = zero, o1 = zero, o2 = zero, o3 = zero;
  float mrow[8], lrow[8];
#pragma unroll
  for (int v = 0; v < 8; v++) { mrow[v] = -1e30f; lrow[v] = 0.f; }

  const int nl = lane & 15;
  const int mh = (lane >> 4) * 8;
  const float scale = 0.125f;                   // 1/sqrt(64)

  for (int r0 = 0; r0 < RG; r0 += 32) {
    // ---- scores: S = Q . K^T for 16 queries x 32 keys ----
    v8f s0 = zero, s1 = zero;
    s0 = wmma_bf16(qf0, load_b_frag(Kb + (size_t)r0 * DH,          DH, lane), s0);
    s0 = wmma_bf16(qf1, load_b_frag(Kb + (size_t)r0 * DH + 32,     DH, lane), s0);
    s1 = wmma_bf16(qf0, load_b_frag(Kb + (size_t)(r0 + 16) * DH,      DH, lane), s1);
    s1 = wmma_bf16(qf1, load_b_frag(Kb + (size_t)(r0 + 16) * DH + 32, DH, lane), s1);

    const float msk0 = mb[r0 + nl]      ? 0.f : -1e30f;
    const float msk1 = mb[r0 + 16 + nl] ? 0.f : -1e30f;

    float al[8];
#pragma unroll
    for (int v = 0; v < 8; v++) {
      float f0 = s0[v] * scale + msk0;
      float f1 = s1[v] * scale + msk1;
      // row max across the 16 N-lanes of this half-wave
      float cm = fmaxf(f0, f1);
      cm = fmaxf(cm, __shfl_xor(cm, 1, 32));
      cm = fmaxf(cm, __shfl_xor(cm, 2, 32));
      cm = fmaxf(cm, __shfl_xor(cm, 4, 32));
      cm = fmaxf(cm, __shfl_xor(cm, 8, 32));
      const float mn = fmaxf(mrow[v], cm);
      const float a  = __expf(mrow[v] - mn);
      const float e0 = __expf(f0 - mn);
      const float e1 = __expf(f1 - mn);
      float rs = e0 + e1;
      rs += __shfl_xor(rs, 1, 32);
      rs += __shfl_xor(rs, 2, 32);
      rs += __shfl_xor(rs, 4, 32);
      rs += __shfl_xor(rs, 8, 32);
      lrow[v] = lrow[v] * a + rs;
      mrow[v] = mn;
      al[v]   = a;
      // stage P (C-layout -> LDS row-major 16x32)
      myP[(v + mh) * 32 + nl]      = f2bf(e0);
      myP[(v + mh) * 32 + 16 + nl] = f2bf(e1);
    }
#pragma unroll
    for (int v = 0; v < 8; v++) {
      o0[v] *= al[v]; o1[v] *= al[v]; o2[v] *= al[v]; o3[v] *= al[v];
    }
    // ---- O += P . V (P as A-fragment from LDS, V^T columns k-contiguous) ----
    v16bf pf = load_a_frag(myP, 32, lane);
    o0 = wmma_bf16(pf, load_b_frag(Vb + (size_t)0  * RG + r0, RG, lane), o0);
    o1 = wmma_bf16(pf, load_b_frag(Vb + (size_t)16 * RG + r0, RG, lane), o1);
    o2 = wmma_bf16(pf, load_b_frag(Vb + (size_t)32 * RG + r0, RG, lane), o2);
    o3 = wmma_bf16(pf, load_b_frag(Vb + (size_t)48 * RG + r0, RG, lane), o3);
  }

  const float av = flags[b] ? 1.f : 0.f;
#pragma unroll
  for (int v = 0; v < 8; v++) {
    const float inv = (lrow[v] > 0.f) ? (av / lrow[v]) : 0.f;
    const size_t row = (size_t)b * MAT + m0 + mh + v;
    unsigned short* op = attnout + row * (NH * DH) + h * DH + nl;
    op[0]  = f2bf(o0[v] * inv);
    op[16] = f2bf(o1[v] * inv);
    op[32] = f2bf(o2[v] * inv);
    op[48] = f2bf(o3[v] * inv);
  }
}

// ---------------------------------------------------------------------------
// Output GEMM + bias + residual: y[m,n] = xn[m,n] + bo[n] + sum_k P[m,k]*Wo[n,k]
// ---------------------------------------------------------------------------
__global__ void out_gemm_kernel(const unsigned short* __restrict__ Abf,
                                const unsigned short* __restrict__ Wbf,
                                const float* __restrict__ bias,
                                float* __restrict__ y) {
  const int lane = threadIdx.x & 31;
  const int wv   = threadIdx.x >> 5;
  const int m0   = blockIdx.y * 64 + wv * 16;
  const int n0   = blockIdx.x * 64;
  v8f zero = {};
  v8f acc[4];
#pragma unroll
  for (int t = 0; t < 4; t++) acc[t] = zero;

  for (int k0 = 0; k0 < FF; k0 += 32) {
    v16bf a = load_a_frag(Abf + (size_t)m0 * FF + k0, FF, lane);
#pragma unroll
    for (int t = 0; t < 4; t++) {
      v16bf bfrag = load_b_frag(Wbf + (size_t)(n0 + t * 16) * FF + k0, FF, lane);
      acc[t] = wmma_bf16(a, bfrag, acc[t]);
    }
  }

  const int nl = lane & 15;
  const int mh = (lane >> 4) * 8;
#pragma unroll
  for (int t = 0; t < 4; t++) {
    const int n = n0 + t * 16 + nl;
    const float bn = bias[n];
#pragma unroll
    for (int v = 0; v < 8; v++) {
      const size_t m = (size_t)m0 + mh + v;
      float* yp = y + m * FF + n;
      *yp = *yp + acc[t][v] + bn;
    }
  }
}

// ---------------------------------------------------------------------------
// Host launch
// ---------------------------------------------------------------------------
extern "C" void kernel_launch(void* const* d_in, const int* in_sizes, int n_in,
                              void* d_out, int out_size, void* d_ws, size_t ws_size,
                              hipStream_t stream) {
  (void)in_sizes; (void)n_in; (void)out_size; (void)ws_size;
  const float* x     = (const float*)d_in[0];
  const float* rg    = (const float*)d_in[1];
  const int*   mask  = (const int*)d_in[2];
  const float* Wq    = (const float*)d_in[3];
  const float* bq    = (const float*)d_in[4];
  const float* Wk    = (const float*)d_in[5];
  const float* bk    = (const float*)d_in[6];
  const float* Wv    = (const float*)d_in[7];
  const float* bv    = (const float*)d_in[8];
  const float* Wo    = (const float*)d_in[9];
  const float* bo    = (const float*)d_in[10];
  const float* gamma = (const float*)d_in[11];
  const float* beta  = (const float*)d_in[12];
  float* y = (float*)d_out;

  char* wp = (char*)d_ws;
  auto alloc = [&](size_t bytes) {
    char* p = wp;
    wp += (bytes + 255) & ~(size_t)255;
    return p;
  };
  unsigned short* xnbf  = (unsigned short*)alloc((size_t)MALL * FF * 2);
  unsigned short* wqb   = (unsigned short*)alloc((size_t)FF * FF * 2);
  unsigned short* wkb   = (unsigned short*)alloc((size_t)FF * FF * 2);
  unsigned short* wvb   = (unsigned short*)alloc((size_t)FF * FF * 2);
  unsigned short* wob   = (unsigned short*)alloc((size_t)FF * FF * 2);
  unsigned short* rgb   = (unsigned short*)alloc((size_t)MKV * FF * 2);
  unsigned short* qb    = (unsigned short*)alloc((size_t)MALL * FF * 2);
  unsigned short* kb    = (unsigned short*)alloc((size_t)MKV * FF * 2);
  unsigned short* vtb   = (unsigned short*)alloc((size_t)MKV * FF * 2);
  unsigned short* attnb = (unsigned short*)alloc((size_t)MALL * FF * 2);
  int* flags = (int*)alloc(2 * sizeof(int));

  // 1) bf16 conversions + mask reduction
  cvt_bf16_kernel<<<128, 256, 0, stream>>>(Wq, wqb, FF * FF);
  cvt_bf16_kernel<<<128, 256, 0, stream>>>(Wk, wkb, FF * FF);
  cvt_bf16_kernel<<<128, 256, 0, stream>>>(Wv, wvb, FF * FF);
  cvt_bf16_kernel<<<128, 256, 0, stream>>>(Wo, wob, FF * FF);
  cvt_bf16_kernel<<<512, 256, 0, stream>>>(rg, rgb, MKV * FF);
  mask_any_kernel<<<BB, 256, 0, stream>>>(mask, flags);

  // 2) LayerNorm (fp32 residual into d_out, bf16 operand into ws)
  ln_kernel<<<MALL, 256, 0, stream>>>(x, gamma, beta, y, xnbf);

  // 3) QKV projections (WMMA)
  proj_gemm_kernel<<<dim3(4, MALL / 64), 128, 0, stream>>>(xnbf, wqb, bq, qb,  MAT, 0);
  proj_gemm_kernel<<<dim3(4, MKV  / 64), 128, 0, stream>>>(rgb,  wkb, bk, kb,  RG,  0);
  proj_gemm_kernel<<<dim3(4, MKV  / 64), 128, 0, stream>>>(rgb,  wvb, bv, vtb, RG,  1);

  // 4) Flash attention (WMMA QK^T + online softmax + WMMA PV)
  attn_kernel<<<dim3(MAT / 64, NH, BB), 128, 0, stream>>>(qb, kb, vtb, mask, flags,
                                                          attnb);

  // 5) Output projection + bias + residual (WMMA)
  out_gemm_kernel<<<dim3(4, MALL / 64), 128, 0, stream>>>(attnb, wob, bo, y);
}
